// Multiheadselfattention_7516192768578
// MI455X (gfx1250) — compile-verified, matrix operand order checked by reading, not run on hardware
//
#include <hip/hip_runtime.h>
#include <hip/hip_bf16.h>
#include <math.h>

typedef __attribute__((ext_vector_type(16))) __bf16 v16bf;
typedef __attribute__((ext_vector_type(8)))  __bf16 v8bf;
typedef __attribute__((ext_vector_type(8)))  float  v8f;
typedef __attribute__((ext_vector_type(2)))  float  v2f;
typedef __attribute__((ext_vector_type(4)))  unsigned int v4u;
typedef __attribute__((ext_vector_type(8)))  int    v8i;
typedef __attribute__((ext_vector_type(4)))  int    v4i;

#define BATCH   4
#define SEQ     1024
#define DIM_IN  512
#define DIM_K   512
#define DIM_V   64
#define NH      64
#define DKH     8                     // DIM_K / NH
#define NTOK    (BATCH*SEQ)           // 4096
#define NQKV    (DIM_K + DIM_K + DIM_V)   // 1088 fused output columns

#define USE_TDM 1                     // stage K/V into LDS with the Tensor Data Mover

#if defined(__has_builtin)
#if __has_builtin(__builtin_amdgcn_wmma_f32_16x16x4_f32)
#define HAVE_WMMA_F32 1
#endif
#endif
#ifndef HAVE_WMMA_F32
#define HAVE_WMMA_F32 0
#endif

// ---------------------------------------------------------------------------
// Kernel 1: x (fp32) -> bf16, row-major [NTOK, DIM_IN]
// ---------------------------------------------------------------------------
__global__ __launch_bounds__(256) void convert_x_kernel(const float* __restrict__ x,
                                                        __bf16* __restrict__ xb) {
    int i = blockIdx.x * 256 + threadIdx.x;
    xb[i] = (__bf16)x[i];
}

// ---------------------------------------------------------------------------
// Kernel 2: fused transposed weight Wt[n][k] (bf16), n in [0,1088), k in [0,512)
// ---------------------------------------------------------------------------
__global__ __launch_bounds__(256) void convert_w_kernel(const float* __restrict__ Wq,
                                                        const float* __restrict__ Wk,
                                                        const float* __restrict__ Wv,
                                                        __bf16* __restrict__ wt) {
    int idx = blockIdx.x * 256 + threadIdx.x;       // over NQKV*DIM_IN
    int n = idx / DIM_IN;
    int k = idx - n * DIM_IN;
    float w;
    if (n < DIM_K)            w = Wq[(size_t)k * DIM_K + n];
    else if (n < 2 * DIM_K)   w = Wk[(size_t)k * DIM_K + (n - DIM_K)];
    else                      w = Wv[(size_t)k * DIM_V + (n - 2 * DIM_K)];
    wt[idx] = (__bf16)w;
}

// ---------------------------------------------------------------------------
// Kernel 3: fused QKV projection GEMM + bias + ReLU using WMMA bf16.
//   C[4096, 1088] = relu(Xb[4096,512] @ Wt^T + bias)
// One 16x16 output tile per wave; 16 v_wmma_f32_16x16x32_bf16 per tile.
// ---------------------------------------------------------------------------
__global__ __launch_bounds__(256) void qkv_gemm_kernel(const __bf16* __restrict__ xb,
                                                       const __bf16* __restrict__ wt,
                                                       const float* __restrict__ bq,
                                                       const float* __restrict__ bk,
                                                       const float* __restrict__ bv,
                                                       float* __restrict__ qo,
                                                       float* __restrict__ ko,
                                                       float* __restrict__ vo) {
    const int lane = threadIdx.x & 31;
    const int wave = threadIdx.x >> 5;
    const int tile = blockIdx.x * 8 + wave;
    const int TN   = NQKV / 16;          // 68 tiles across N
    const int tm   = tile / TN;
    const int tn   = tile - tm * TN;

    const int hsel = lane >> 4;          // 0: K base 0, 1: K base 8
    const int l15  = lane & 15;

    const __bf16* aptr = xb + (size_t)(tm * 16 + l15) * DIM_IN + hsel * 8;
    const __bf16* bptr = wt + (size_t)(tn * 16 + l15) * DIM_IN + hsel * 8;

    v8f c = {};
#pragma unroll
    for (int kb = 0; kb < DIM_IN; kb += 32) {
        v8bf a0 = *(const v8bf*)(aptr + kb);
        v8bf a1 = *(const v8bf*)(aptr + kb + 16);
        v8bf b0 = *(const v8bf*)(bptr + kb);
        v8bf b1 = *(const v8bf*)(bptr + kb + 16);
        v16bf A, B;
#pragma unroll
        for (int i = 0; i < 8; ++i) {
            A[i] = a0[i];  A[i + 8] = a1[i];
            B[i] = b0[i];  B[i + 8] = b1[i];
        }
        c = __builtin_amdgcn_wmma_f32_16x16x32_bf16(
                /*neg_a=*/false, A, /*neg_b=*/false, B,
                /*c_mod=*/(short)0, c, /*reuse_a=*/false, /*reuse_b=*/false);
    }

    // C layout: VGPR r -> row tm*16 + r + hsel*8, col tn*16 + (lane&15)
    const int col  = tn * 16 + l15;
    const float bias = (col < DIM_K)     ? bq[col]
                     : (col < 2 * DIM_K) ? bk[col - DIM_K]
                                         : bv[col - 2 * DIM_K];
#pragma unroll
    for (int r = 0; r < 8; ++r) {
        const int row = tm * 16 + r + hsel * 8;
        const float val = fmaxf(c[r] + bias, 0.0f);
        if (col < DIM_K)          qo[(size_t)row * DIM_K + col]              = val;
        else if (col < 2 * DIM_K) ko[(size_t)row * DIM_K + (col - DIM_K)]   = val;
        else                      vo[(size_t)row * DIM_V + (col - 2*DIM_K)] = val;
    }
}

// ---------------------------------------------------------------------------
// TDM helper: DMA a 2D fp32 tile from a strided global tensor into LDS.
// Descriptor packing per CDNA5 ISA §8.3/8.4; groups 2/3 zero (2D tensor).
// ---------------------------------------------------------------------------
#if USE_TDM
__device__ __forceinline__ void tdm_load_2d_f32(unsigned lds_addr, const void* gaddr,
                                                unsigned tensor_d0, unsigned tensor_d1,
                                                unsigned tile_d0, unsigned tile_d1,
                                                unsigned stride_d0) {
    const unsigned long long ga = (unsigned long long)gaddr;
    v4u g0;
    g0[0] = 1u;                                            // count=1, user mode
    g0[1] = lds_addr;                                      // LDS byte address
    g0[2] = (unsigned)(ga & 0xFFFFFFFFu);                  // global_addr[31:0]
    g0[3] = (unsigned)((ga >> 32) & 0x01FFFFFFu) | (2u << 30); // addr[56:32], type=2

    v8i g1;
    g1[0] = (int)(2u << 16);                               // data_size=2 (4B), mask=0
    g1[1] = (int)((tensor_d0 & 0xFFFFu) << 16);            // tensor_dim0[15:0]
    g1[2] = (int)(((tensor_d0 >> 16) & 0xFFFFu) | ((tensor_d1 & 0xFFFFu) << 16));
    g1[3] = (int)(((tensor_d1 >> 16) & 0xFFFFu) | ((tile_d0 & 0xFFFFu) << 16));
    g1[4] = (int)(tile_d1 & 0xFFFFu);                      // tile_dim1, tile_dim2=0
    g1[5] = (int)stride_d0;                                // tensor_dim0_stride[31:0]
    g1[6] = 0;
    g1[7] = 0;

    v4i z4 = {0, 0, 0, 0};
#if __clang_major__ >= 23
    v8i z8 = {0, 0, 0, 0, 0, 0, 0, 0};
    __builtin_amdgcn_tensor_load_to_lds(g0, g1, z4, z4, z8, 0);
#else
    __builtin_amdgcn_tensor_load_to_lds(g0, g1, z4, z4, 0);
#endif
}
#endif

// ---------------------------------------------------------------------------
// Kernel 4: attention per (batch, head), scores on the f32 WMMA path.
// K slab (32KB) + V (4KB) LDS-resident via TDM. Each wave owns 16-query
// tiles; a 16x16 score tile = two v_wmma_f32_16x16x4_f32 (dk=8). Row stats
// live per-lane in the C layout and are reduced with 16-lane-half shuffles.
// ---------------------------------------------------------------------------
__global__ __launch_bounds__(256) void attention_kernel(const float* __restrict__ qg,
                                                        const float* __restrict__ kg,
                                                        const float* __restrict__ vg,
                                                        float* __restrict__ out) {
    __shared__ float Ks[SEQ * DKH];   // 32 KB
    __shared__ float Vs[SEQ];         //  4 KB

    const int bh   = blockIdx.x;      // 0..255
    const int b    = bh >> 6;
    const int h    = bh & 63;
    const int t    = threadIdx.x;
    const int lane = t & 31;
    const int wave = t >> 5;
    const int l15  = lane & 15;
    const int hsel = lane >> 4;       // C layout: rows r (half 0) / r+8 (half 1)

#if USE_TDM
    if (t < 32) {   // wave 0 issues both DMA descriptors (TDM ignores EXEC)
        const unsigned ksLds = (unsigned)(unsigned long long)(void*)Ks;
        const unsigned vsLds = (unsigned)(unsigned long long)(void*)Vs;
        tdm_load_2d_f32(ksLds, kg + ((size_t)b * SEQ) * DIM_K + h * DKH,
                        DIM_K, NTOK, DKH, SEQ, DIM_K);
        tdm_load_2d_f32(vsLds, vg + ((size_t)b * SEQ) * DIM_V + h,
                        DIM_V, NTOK, 1, SEQ, DIM_V);
        __builtin_amdgcn_s_wait_tensorcnt(0);
    }
#else
    for (int i = t; i < SEQ; i += 256) {
        const float* kr = kg + ((size_t)(b * SEQ + i)) * DIM_K + h * DKH;
        *(float4*)(Ks + i * DKH)     = *(const float4*)kr;
        *(float4*)(Ks + i * DKH + 4) = *(const float4*)(kr + 4);
        Vs[i] = vg[((size_t)(b * SEQ + i)) * DIM_V + h];
    }
#endif
    __syncthreads();

    const float scale = 0.35355339059327373f;            // 1/sqrt(8)
    const int   fe    = (h < 32) ? h : (h - 32);
    const float invf  = __expf(-(float)(2 * fe) * (1.0f / 64.0f) * 9.210340371976184f);

#if HAVE_WMMA_F32
    for (int qt = wave; qt < SEQ / 16; qt += 8) {
        // A-fragments (32-bit A 16x4 layout): lane row = l15, K pair by hsel
        const float* qr = qg + ((size_t)(b * SEQ + qt * 16 + l15)) * DIM_K
                             + h * DKH + 2 * hsel;
        const v2f A0 = { qr[0], qr[1] };   // K = 0..3
        const v2f A1 = { qr[4], qr[5] };   // K = 4..7

        v8f mloc;
#pragma unroll
        for (int r = 0; r < 8; ++r) mloc[r] = -3.0e38f;

        // ---- pass 1: per-lane per-row max over this lane's 64 key columns
        for (int kt = 0; kt < SEQ / 16; ++kt) {
            const float* kr = Ks + (kt * 16 + l15) * DKH + 2 * hsel;
            const v2f B0 = { kr[0], kr[1] };
            const v2f B1 = { kr[4], kr[5] };
            v8f c = {};
            c = __builtin_amdgcn_wmma_f32_16x16x4_f32(false, A0, false, B0,
                                                      (short)0, c, false, false);
            c = __builtin_amdgcn_wmma_f32_16x16x4_f32(false, A1, false, B1,
                                                      (short)0, c, false, false);
#pragma unroll
            for (int r = 0; r < 8; ++r) mloc[r] = fmaxf(mloc[r], c[r]);
        }
        // reduce row max across the 16 lanes of each half (masks stay in-half)
#pragma unroll
        for (int off = 1; off < 16; off <<= 1)
#pragma unroll
            for (int r = 0; r < 8; ++r)
                mloc[r] = fmaxf(mloc[r], __shfl_xor(mloc[r], off, 32));

        // ---- pass 2: exp-sum and weighted V accumulation
        v8f lloc = {}, acc = {};
        for (int kt = 0; kt < SEQ / 16; ++kt) {
            const float* kr = Ks + (kt * 16 + l15) * DKH + 2 * hsel;
            const v2f B0 = { kr[0], kr[1] };
            const v2f B1 = { kr[4], kr[5] };
            const float vv = Vs[kt * 16 + l15];
            v8f c = {};
            c = __builtin_amdgcn_wmma_f32_16x16x4_f32(false, A0, false, B0,
                                                      (short)0, c, false, false);
            c = __builtin_amdgcn_wmma_f32_16x16x4_f32(false, A1, false, B1,
                                                      (short)0, c, false, false);
#pragma unroll
            for (int r = 0; r < 8; ++r) {
                const float p = __expf((c[r] - mloc[r]) * scale);
                lloc[r] += p;
                acc[r]  += p * vv;
            }
        }
#pragma unroll
        for (int off = 1; off < 16; off <<= 1)
#pragma unroll
            for (int r = 0; r < 8; ++r) {
                lloc[r] += __shfl_xor(lloc[r], off, 32);
                acc[r]  += __shfl_xor(acc[r],  off, 32);
            }

        // one lane per half writes its row, fused with position embedding
#pragma unroll
        for (int r = 0; r < 8; ++r) {
            if (l15 == r) {
                const int   i   = qt * 16 + r + hsel * 8;
                const float ang = (float)i * invf;
                const float pe  = (h < 32) ? __cosf(ang) : __sinf(ang);
                out[((size_t)(b * SEQ + i)) * DIM_V + h] = acc[r] / lloc[r] + pe;
            }
        }
    }
#else   // ------- VALU fallback (two-pass scalar softmax) -------
    for (int i = t; i < SEQ; i += 256) {
        const float* qr = qg + ((size_t)(b * SEQ + i)) * DIM_K + h * DKH;
        const float4 qa  = *(const float4*)qr;
        const float4 qb4 = *(const float4*)(qr + 4);
        float m = -3.0e38f;
        for (int kk = 0; kk < SEQ; ++kk) {
            const float4 ka = *(const float4*)(Ks + kk * DKH);
            const float4 kb = *(const float4*)(Ks + kk * DKH + 4);
            float s = qa.x*ka.x + qa.y*ka.y + qa.z*ka.z + qa.w*ka.w
                    + qb4.x*kb.x + qb4.y*kb.y + qb4.z*kb.z + qb4.w*kb.w;
            m = fmaxf(m, s);
        }
        float l = 0.0f, av = 0.0f;
        for (int kk = 0; kk < SEQ; ++kk) {
            const float4 ka = *(const float4*)(Ks + kk * DKH);
            const float4 kb = *(const float4*)(Ks + kk * DKH + 4);
            float s = qa.x*ka.x + qa.y*ka.y + qa.z*ka.z + qa.w*ka.w
                    + qb4.x*kb.x + qb4.y*kb.y + qb4.z*kb.z + qb4.w*kb.w;
            float p = __expf((s - m) * scale);
            l += p;  av += p * Vs[kk];
        }
        const float ang = (float)i * invf;
        const float pe  = (h < 32) ? __cosf(ang) : __sinf(ang);
        out[((size_t)(b * SEQ + i)) * DIM_V + h] = av / l + pe;
    }
#endif
}

// ---------------------------------------------------------------------------
// Host-side launcher
// ---------------------------------------------------------------------------
extern "C" void kernel_launch(void* const* d_in, const int* in_sizes, int n_in,
                              void* d_out, int out_size, void* d_ws, size_t ws_size,
                              hipStream_t stream) {
    const float* x  = (const float*)d_in[0];
    const float* Wq = (const float*)d_in[1];
    const float* bq = (const float*)d_in[2];
    const float* Wk = (const float*)d_in[3];
    const float* bk = (const float*)d_in[4];
    const float* Wv = (const float*)d_in[5];
    const float* bv = (const float*)d_in[6];
    float* out = (float*)d_out;

    char* ws = (char*)d_ws;
    size_t off = 0;
    __bf16* xb = (__bf16*)(ws + off);  off += (size_t)NTOK * DIM_IN * sizeof(__bf16);  // 4 MiB
    __bf16* wt = (__bf16*)(ws + off);  off += (size_t)NQKV * DIM_IN * sizeof(__bf16);  // ~1.06 MiB
    float*  qb = (float*)(ws + off);   off += (size_t)NTOK * DIM_K * sizeof(float);    // 8 MiB
    float*  kb = (float*)(ws + off);   off += (size_t)NTOK * DIM_K * sizeof(float);    // 8 MiB
    float*  vb = (float*)(ws + off);   off += (size_t)NTOK * DIM_V * sizeof(float);    // 1 MiB

    convert_x_kernel<<<(NTOK * DIM_IN) / 256, 256, 0, stream>>>(x, xb);
    convert_w_kernel<<<(NQKV * DIM_IN) / 256, 256, 0, stream>>>(Wq, Wk, Wv, wt);

    const int n_tiles = (NTOK / 16) * (NQKV / 16);   // 256 * 68 = 17408
    qkv_gemm_kernel<<<n_tiles / 8, 256, 0, stream>>>(xb, wt, bq, bk, bv, qb, kb, vb);

    attention_kernel<<<BATCH * NH, 256, 0, stream>>>(qb, kb, vb, out);
}
